// EdfExtractorLight_70257075028697
// MI455X (gfx1250) — matrix operand order repaired and checked by reading, not compiled, not run on hardware
//
#include <hip/hip_runtime.h>
#include <hip/hip_bf16.h>
#include <math.h>

#define NQd   4096
#define NNODE 40960
#define FDIM  128
#define HH    4
#define HDd   32
#define NB64  64
#define EEDGE 262144
#define E0CNT 131072

typedef __attribute__((ext_vector_type(16))) __bf16 v16bf;
typedef __attribute__((ext_vector_type(8)))  float  v8f;
typedef __attribute__((ext_vector_type(4)))  unsigned int v4u;
typedef __attribute__((ext_vector_type(8)))  int v8i;
typedef __attribute__((ext_vector_type(4)))  int v4i;

__device__ __forceinline__ v8f wmma_bf16(v16bf a, v16bf b, v8f c) {
  return __builtin_amdgcn_wmma_f32_16x16x32_bf16(false, a, false, b, (short)0, c, false, false);
}

// ---- Tensor Data Mover: async 2D f32 tile load (global -> LDS), TENSORcnt-tracked.
// Descriptor layout per CDNA5 ISA:
//  g0: [1:0]=count=1, [63:32]=lds_addr, [120:64]=global_addr, [127:126]=type=2
//  g1: [17:16]=data_size(2=4B), [79:48]=tensor_dim0, [111:80]=tensor_dim1,
//      [127:112]=tile_dim0, [143:128]=tile_dim1, [207:160]=tensor_dim0_stride
__device__ __forceinline__ void tdm_load_2d_f32(const void* gaddr, unsigned lds_off,
                                                unsigned tensor_d0, unsigned tensor_d1,
                                                unsigned tile_d0, unsigned tile_d1,
                                                unsigned stride0) {
  unsigned long long ga = (unsigned long long)(size_t)gaddr;
  v4u g0;
  g0[0] = 1u;                                                  // count=1 (valid user desc)
  g0[1] = lds_off;                                             // lds_addr (bytes)
  g0[2] = (unsigned)(ga & 0xffffffffu);                        // global_addr[31:0]
  g0[3] = (unsigned)((ga >> 32) & 0x01ffffffu) | 0x80000000u;  // addr[56:32] | type=2
  v8i g1;
  g1[0] = (int)(2u << 16);                                     // data_size = 4 bytes
  g1[1] = (int)(tensor_d0 << 16);                              // tensor_dim0[15:0]
  g1[2] = (int)((tensor_d0 >> 16) | (tensor_d1 << 16));        // d0[31:16] | d1[15:0]
  g1[3] = (int)((tensor_d1 >> 16) | (tile_d0 << 16));          // d1[31:16] | tile_dim0
  g1[4] = (int)(tile_d1 & 0xffffu);                            // tile_dim1 (tile_dim2=0)
  g1[5] = (int)stride0;                                        // tensor_dim0_stride[31:0]
  g1[6] = 0;
  g1[7] = 0;
  v4i z4 = {0, 0, 0, 0};
#if defined(__clang_major__) && (__clang_major__ >= 23)
  v8i z8 = {0, 0, 0, 0, 0, 0, 0, 0};
  __builtin_amdgcn_tensor_load_to_lds(g0, g1, z4, z4, z8, 0);
#else
  __builtin_amdgcn_tensor_load_to_lds(g0, g1, z4, z4, 0);
#endif
}

__device__ __forceinline__ unsigned lds_offset_of(const void* p) {
  // LDS aperture maps generic addr[31:0] to the LDS offset (ISA §10.2).
  return (unsigned)(size_t)p;
}

// A-fragment (16x32 bf16, MxK) from row-major LDS tile [rows][ld]
__device__ __forceinline__ v16bf load_a_frag(const __bf16* a, int ld, int lane, int k0) {
  const int row = lane & 15, half = lane >> 4;
  v16bf f;
#pragma unroll
  for (int i = 0; i < 16; ++i) {
    int k = k0 + (i & 7) + half * 8 + ((i >> 3) << 4);
    f[i] = a[row * ld + k];
  }
  return f;
}

// B-fragment (32x16 bf16, KxN) from K-contiguous transposed LDS tile BT[col][ldk]
__device__ __forceinline__ v16bf load_b_frag(const __bf16* bt, int ldk, int lane, int n0, int k0) {
  const int col = n0 + (lane & 15), half = lane >> 4;
  const __bf16* p = bt + col * ldk + k0 + half * 8;
  v16bf f;
#pragma unroll
  for (int i = 0; i < 16; ++i) f[i] = p[(i & 7) + ((i >> 3) << 4)];
  return f;
}

// Gaussian radial-basis A-fragment built directly in registers.
__device__ __forceinline__ v16bf g_frag(float xnv, float sfv, int half, int k0) {
  v16bf f;
#pragma unroll
  for (int i = 0; i < 16; ++i) {
    int k = k0 + (i & 7) + half * 8 + ((i >> 3) << 4);
    float d = (xnv - (float)k * (1.0f / 63.0f)) * 64.0f;
    f[i] = (__bf16)(__expf(-0.5f * d * d) * sfv);
  }
  return f;
}

// ---------------- k_prep: q = b_dst @ Wq ; Wkq[f][h] = sum_d Wk[f,h*32+d]*q[h*32+d]
__global__ void __launch_bounds__(128) k_prep(const float* __restrict__ b_dst,
                                              const float* __restrict__ Wq,
                                              const float* __restrict__ Wk,
                                              float* __restrict__ wkq) {
  __shared__ float qs[FDIM];
  const int d = threadIdx.x;
  float acc = 0.f;
  for (int f = 0; f < FDIM; ++f) acc += b_dst[f] * Wq[f * FDIM + d];
  qs[d] = acc;
  __syncthreads();
  const int f = d;
  for (int h = 0; h < 16; ++h) {
    float a = 0.f;
    if (h < HH) {
      for (int j = 0; j < HDd; ++j) a += Wk[f * FDIM + h * HDd + j] * qs[h * HDd + j];
    }
    wkq[f * 16 + h] = a;  // cols 4..15 zero-padded
  }
}

// ---------------- k_init
__global__ void k_init(float* __restrict__ m, float* __restrict__ den, float* __restrict__ acc) {
  int i = blockIdx.x * blockDim.x + threadIdx.x;
  if (i < NQd * HH) { m[i] = -3.0e38f; den[i] = 0.f; }
  if (i < NQd * FDIM) acc[i] = 0.f;
}

// ---------------- k_project: Vf = node_feature @ Wv ; node_logit = node_feature @ Wkq
__global__ void __launch_bounds__(256) k_project(const float* __restrict__ nodef,
                                                 const float* __restrict__ Wv,
                                                 const float* __restrict__ wkq,
                                                 float* __restrict__ Vf,
                                                 float* __restrict__ nlog) {
  __shared__ float  sStageA[16 * 128];  // TDM target (f32)
  __shared__ __bf16 sA[16 * 128];       // node rows, bf16
  __shared__ __bf16 sBT[144 * 128];     // [col][k], cols 0..127 = Wv, 128..143 = Wkq
  const int tid = threadIdx.x;
  const int r0 = blockIdx.x * 16;
  // Async DMA of the 16x128 node tile into LDS while we stage weights.
  if (tid < 32) {
    tdm_load_2d_f32(nodef + (size_t)r0 * FDIM, lds_offset_of(sStageA),
                    FDIM, NNODE, FDIM, 16, FDIM);
  }
  {
    int f = tid >> 1, c0 = (tid & 1) * 64;
    const float* wrow = Wv + f * FDIM + c0;
#pragma unroll 4
    for (int c = 0; c < 64; ++c) sBT[(c0 + c) * 128 + f] = (__bf16)wrow[c];
    int cc0 = (tid & 1) * 8;
    const float* krow = wkq + f * 16 + cc0;
#pragma unroll
    for (int c = 0; c < 8; ++c) sBT[(128 + cc0 + c) * 128 + f] = (__bf16)krow[c];
  }
  if (tid < 32) __builtin_amdgcn_s_wait_tensorcnt(0);
  __syncthreads();
  {
#pragma unroll
    for (int i = 0; i < 8; ++i) sA[tid * 8 + i] = (__bf16)sStageA[tid * 8 + i];
  }
  __syncthreads();
  const int wave = tid >> 5, lane = tid & 31;
  const int half = lane >> 4, l15 = lane & 15;
  {
    v8f c = {};
#pragma unroll
    for (int k0 = 0; k0 < 128; k0 += 32)
      c = wmma_bf16(load_a_frag(sA, 128, lane, k0), load_b_frag(sBT, 128, lane, wave * 16, k0), c);
#pragma unroll
    for (int j = 0; j < 8; ++j)
      Vf[(r0 + j + half * 8) * FDIM + wave * 16 + l15] = c[j];
  }
  if (wave == 0) {  // uniform per wave -> EXEC all ones inside
    v8f c = {};
#pragma unroll
    for (int k0 = 0; k0 < 128; k0 += 32)
      c = wmma_bf16(load_a_frag(sA, 128, lane, k0), load_b_frag(sBT, 128, lane, 128, k0), c);
    if (l15 < HH) {
#pragma unroll
      for (int j = 0; j < 8; ++j)
        nlog[(r0 + j + half * 8) * HH + l15] = c[j];
    }
  }
}

// ---------------- k_edges: edge_scalar via WMMA, logits, atomic segment max
__global__ void __launch_bounds__(256) k_edges(const float* __restrict__ qc,
                                               const float* __restrict__ ncrd,
                                               const int* __restrict__ esrc,
                                               const int* __restrict__ edst,
                                               const int* __restrict__ sid,
                                               const float* __restrict__ Wr0,
                                               const float* __restrict__ br0,
                                               const float* __restrict__ Wr1,
                                               const float* __restrict__ br1,
                                               const float* __restrict__ Wal,
                                               const float* __restrict__ nlog,
                                               float* __restrict__ logits,
                                               float* __restrict__ mbuf) {
  __shared__ float sStageW[64 * 64];  // TDM target: W_rad f32
  __shared__ float sXn[128], sSoft[128];
  __shared__ int   sDst[128];
  __shared__ float sNl[128 * 4];
  __shared__ __bf16 sWT[64 * 64];   // W_rad^T [col][k]
  __shared__ float sBr[64];
  __shared__ float sWa[64 * 4];
  __shared__ float sEs[8 * 16 * 64];
  const int tid = threadIdx.x;
  const int e0 = blockIdx.x * 128;
  const int scale = sid[e0];  // uniform per 128-edge block by construction
  const float* Wr = scale ? Wr1 : Wr0;
  const float* br = scale ? br1 : br0;
  if (tid < 32) {
    tdm_load_2d_f32(Wr, lds_offset_of(sStageW), NB64, NB64, NB64, NB64, NB64);
  }
  if (tid < 128) {
    int e = e0 + tid;
    int src = esrc[e], dst = edst[e];
    float vx = ncrd[src * 3 + 0] - qc[dst * 3 + 0];
    float vy = ncrd[src * 3 + 1] - qc[dst * 3 + 1];
    float vz = ncrd[src * 3 + 2] - qc[dst * 3 + 2];
    float rl = sqrtf(vx * vx + vy * vy + vz * vz + 1e-12f);
    float off = scale ? 0.2f : 0.1f;
    float cut = scale ? 4.0f : 2.0f;
    float xn = (rl - off) / (cut - off);
    float tc = fminf(fmaxf((xn - 0.8f) * 5.0f, 0.f), 1.f);
    sXn[tid] = xn;
    sSoft[tid] = 0.5f * (1.0f + __cosf(3.14159265358979f * tc));
    sDst[tid] = dst;
#pragma unroll
    for (int h = 0; h < 4; ++h) sNl[tid * 4 + h] = nlog[src * 4 + h];
  }
  if (tid < 64) sBr[tid] = br[tid];
  sWa[tid] = Wal[tid];
  if (tid < 32) __builtin_amdgcn_s_wait_tensorcnt(0);
  __syncthreads();
  {  // convert + transpose staged W_rad to bf16
    int f = tid >> 2, c0 = (tid & 3) * 16;
#pragma unroll
    for (int c = 0; c < 16; ++c) sWT[(c0 + c) * 64 + f] = (__bf16)sStageW[f * 64 + c0 + c];
  }
  __syncthreads();
  const int wave = tid >> 5, lane = tid & 31;
  const int half = lane >> 4, l15 = lane & 15;
  const int eb = wave * 16;
  {
    float xnv = sXn[eb + l15], sfv = sSoft[eb + l15];
    v16bf a0 = g_frag(xnv, sfv, half, 0);
    v16bf a1 = g_frag(xnv, sfv, half, 32);
#pragma unroll
    for (int t = 0; t < 4; ++t) {
      v8f c = {};
      c = wmma_bf16(a0, load_b_frag(sWT, 64, lane, t * 16, 0), c);
      c = wmma_bf16(a1, load_b_frag(sWT, 64, lane, t * 16, 32), c);
      float bias = sBr[t * 16 + l15];
#pragma unroll
      for (int j = 0; j < 8; ++j)
        sEs[(eb + j + half * 8) * 64 + t * 16 + l15] = c[j] + bias;
    }
  }
  __syncthreads();
#pragma unroll
  for (int it = 0; it < 2; ++it) {
    int me = l15, h = half + 2 * it;
    const float* esrow = sEs + (eb + me) * 64;
    float dot = 0.f;
#pragma unroll 8
    for (int k = 0; k < 64; ++k) dot += esrow[k] * sWa[k * 4 + h];
    float lg = sNl[(eb + me) * 4 + h] * 0.17677669529f + dot;  // 1/sqrt(32)
    logits[(e0 + eb + me) * 4 + h] = lg;
    float* maddr = mbuf + sDst[eb + me] * 4 + h;
    if (lg >= 0.f) atomicMax((int*)maddr, __float_as_int(lg));
    else           atomicMin((unsigned int*)maddr, __float_as_uint(lg));
  }
}

// ---------------- k_soft: p = exp(l - m[dst]); denom += p (in-place into logits buf)
__global__ void k_soft(const int* __restrict__ edst, const float* __restrict__ mbuf,
                       float* __restrict__ pl, float* __restrict__ den) {
  int i = blockIdx.x * blockDim.x + threadIdx.x;
  if (i >= EEDGE * HH) return;
  int e = i >> 2, h = i & 3;
  int dst = edst[e];
  float p = __expf(pl[i] - mbuf[dst * 4 + h]);
  pl[i] = p;
  atomicAdd(den + dst * 4 + h, p);
}

// ---------------- k_msg: recompute edge_scalar, v_e via WMMA, scatter-accumulate
__global__ void __launch_bounds__(256) k_msg(const float* __restrict__ qc,
                                             const float* __restrict__ ncrd,
                                             const int* __restrict__ esrc,
                                             const int* __restrict__ edst,
                                             const int* __restrict__ sid,
                                             const float* __restrict__ Wr0,
                                             const float* __restrict__ br0,
                                             const float* __restrict__ Wr1,
                                             const float* __restrict__ br1,
                                             const float* __restrict__ Wmsg,
                                             const float* __restrict__ Wsh,
                                             const float* __restrict__ Vf,
                                             const float* __restrict__ pl,
                                             const float* __restrict__ den,
                                             float* __restrict__ acc) {
  __shared__ float sXn[128], sSoft[128];
  __shared__ int   sSrc[128], sDst[128];
  __shared__ float sAl[128 * 4];
  __shared__ float sSh[128 * 9];
  __shared__ __bf16 sWTr[64 * 64];    // W_rad^T
  __shared__ float  sBr[64];
  __shared__ __bf16 sWTm[128 * 64];   // W_msg^T [col][k]
  __shared__ __bf16 sWTs[128 * 32];   // W_sh^T zero-padded to K=32
  __shared__ __bf16 sEs[8 * 16 * 64]; // edge_scalar per wave (A of next GEMM)
  const int tid = threadIdx.x;
  const int e0 = blockIdx.x * 128;
  const int scale = sid[e0];
  const float* Wr = scale ? Wr1 : Wr0;
  const float* br = scale ? br1 : br0;
  if (tid < 128) {
    int e = e0 + tid;
    int src = esrc[e], dst = edst[e];
    __builtin_prefetch(Vf + (size_t)src * FDIM, 0, 1);  // global_prefetch of gather row
    float vx = ncrd[src * 3 + 0] - qc[dst * 3 + 0];
    float vy = ncrd[src * 3 + 1] - qc[dst * 3 + 1];
    float vz = ncrd[src * 3 + 2] - qc[dst * 3 + 2];
    float rl = sqrtf(vx * vx + vy * vy + vz * vz + 1e-12f);
    float off = scale ? 0.2f : 0.1f;
    float cut = scale ? 4.0f : 2.0f;
    float xn = (rl - off) / (cut - off);
    float tc = fminf(fmaxf((xn - 0.8f) * 5.0f, 0.f), 1.f);
    sXn[tid] = xn;
    sSoft[tid] = 0.5f * (1.0f + __cosf(3.14159265358979f * tc));
    sSrc[tid] = src; sDst[tid] = dst;
    float inv = 1.0f / rl;
    float ux = vx * inv, uy = vy * inv, uz = vz * inv;
    const float s3 = 1.73205080757f, s15 = 3.87298334621f, s5 = 2.2360679775f;
    float* sh = sSh + tid * 9;
    sh[0] = 1.0f; sh[1] = s3 * ux; sh[2] = s3 * uy; sh[3] = s3 * uz;
    sh[4] = s15 * ux * uy; sh[5] = s15 * uy * uz;
    sh[6] = 0.5f * s5 * (3.0f * uz * uz - 1.0f);
    sh[7] = s15 * ux * uz; sh[8] = 0.5f * s15 * (ux * ux - uy * uy);
#pragma unroll
    for (int h = 0; h < 4; ++h)
      sAl[tid * 4 + h] = pl[e * 4 + h] / (den[dst * 4 + h] + 1e-9f);
  }
  {
    int f = tid >> 2, c0 = (tid & 3) * 16;
#pragma unroll
    for (int c = 0; c < 16; ++c) sWTr[(c0 + c) * 64 + f] = (__bf16)Wr[f * 64 + c0 + c];
  }
  {
    int k = tid >> 2, c0 = (tid & 3) * 32;
#pragma unroll 8
    for (int c = 0; c < 32; ++c) sWTm[(c0 + c) * 64 + k] = (__bf16)Wmsg[k * 128 + c0 + c];
  }
  {
    int col = tid >> 1, kk0 = (tid & 1) * 16;
#pragma unroll
    for (int k = 0; k < 16; ++k) {
      int kk = kk0 + k;
      sWTs[col * 32 + kk] = (kk < 9) ? (__bf16)Wsh[kk * 128 + col] : (__bf16)0.0f;
    }
  }
  if (tid < 64) sBr[tid] = br[tid];
  __syncthreads();
  const int wave = tid >> 5, lane = tid & 31;
  const int half = lane >> 4, l15 = lane & 15;
  const int eb = wave * 16;
  {
    float xnv = sXn[eb + l15], sfv = sSoft[eb + l15];
    v16bf a0 = g_frag(xnv, sfv, half, 0);
    v16bf a1 = g_frag(xnv, sfv, half, 32);
#pragma unroll
    for (int t = 0; t < 4; ++t) {
      v8f c = {};
      c = wmma_bf16(a0, load_b_frag(sWTr, 64, lane, t * 16, 0), c);
      c = wmma_bf16(a1, load_b_frag(sWTr, 64, lane, t * 16, 32), c);
      float bias = sBr[t * 16 + l15];
#pragma unroll
      for (int j = 0; j < 8; ++j)
        sEs[(eb + j + half * 8) * 64 + t * 16 + l15] = (__bf16)(c[j] + bias);
    }
  }
  __syncthreads();
  v16bf aes0 = load_a_frag(sEs + eb * 64, 64, lane, 0);
  v16bf aes1 = load_a_frag(sEs + eb * 64, 64, lane, 32);
  v16bf ash;
  {
    const float* shrow = sSh + (eb + l15) * 9;
#pragma unroll
    for (int i = 0; i < 16; ++i) {
      int k = (i & 7) + half * 8 + ((i >> 3) << 4);
      ash[i] = (k < 9) ? (__bf16)shrow[k] : (__bf16)0.0f;
    }
  }
#pragma unroll
  for (int t = 0; t < 8; ++t) {
    v8f m1 = {};
    m1 = wmma_bf16(aes0, load_b_frag(sWTm, 64, lane, t * 16, 0), m1);
    m1 = wmma_bf16(aes1, load_b_frag(sWTm, 64, lane, t * 16, 32), m1);
    v8f m2 = {};
    m2 = wmma_bf16(ash, load_b_frag(sWTs, 32, lane, t * 16, 0), m2);
    int fcol = t * 16 + l15;
    int h = fcol >> 5;
#pragma unroll
    for (int j = 0; j < 8; ++j) {
      int el = eb + j + half * 8;
      float vf = Vf[sSrc[el] * FDIM + fcol];
      float val = sAl[el * 4 + h] * (vf * m1[j] + m2[j]);
      atomicAdd(acc + sDst[el] * FDIM + fcol, val);
    }
  }
}

// ---------------- k_out: out = acc @ Wo + bo
__global__ void __launch_bounds__(256) k_out(const float* __restrict__ acc,
                                             const float* __restrict__ Wo,
                                             const float* __restrict__ bo,
                                             float* __restrict__ out) {
  __shared__ float  sStageA[16 * 128];  // TDM target (f32)
  __shared__ __bf16 sA[16 * 128];
  __shared__ __bf16 sBT[128 * 128];
  __shared__ float  sBo[128];
  const int tid = threadIdx.x;
  const int r0 = blockIdx.x * 16;
  if (tid < 32) {
    tdm_load_2d_f32(acc + (size_t)r0 * FDIM, lds_offset_of(sStageA),
                    FDIM, NQd, FDIM, 16, FDIM);
  }
  {
    int f = tid >> 1, c0 = (tid & 1) * 64;
    const float* wrow = Wo + f * FDIM + c0;
#pragma unroll 4
    for (int c = 0; c < 64; ++c) sBT[(c0 + c) * 128 + f] = (__bf16)wrow[c];
  }
  if (tid < 128) sBo[tid] = bo[tid];
  if (tid < 32) __builtin_amdgcn_s_wait_tensorcnt(0);
  __syncthreads();
  {
#pragma unroll
    for (int i = 0; i < 8; ++i) sA[tid * 8 + i] = (__bf16)sStageA[tid * 8 + i];
  }
  __syncthreads();
  const int wave = tid >> 5, lane = tid & 31;
  const int half = lane >> 4, l15 = lane & 15;
  v8f c = {};
#pragma unroll
  for (int k0 = 0; k0 < 128; k0 += 32)
    c = wmma_bf16(load_a_frag(sA, 128, lane, k0), load_b_frag(sBT, 128, lane, wave * 16, k0), c);
  float b = sBo[wave * 16 + l15];
#pragma unroll
  for (int j = 0; j < 8; ++j)
    out[(r0 + j + half * 8) * FDIM + wave * 16 + l15] = c[j] + b;
}

extern "C" void kernel_launch(void* const* d_in, const int* in_sizes, int n_in,
                              void* d_out, int out_size, void* d_ws, size_t ws_size,
                              hipStream_t stream) {
  (void)in_sizes; (void)n_in; (void)out_size; (void)ws_size;
  const float* query_coord = (const float*)d_in[0];
  const float* node_feature= (const float*)d_in[1];
  const float* node_coord  = (const float*)d_in[2];
  const int*   edge_src    = (const int*)d_in[3];
  const int*   edge_dst    = (const int*)d_in[4];
  const int*   scale_id    = (const int*)d_in[5];
  const float* b_dst       = (const float*)d_in[6];
  const float* Wq          = (const float*)d_in[7];
  const float* Wk          = (const float*)d_in[8];
  const float* Wv          = (const float*)d_in[9];
  const float* Wo          = (const float*)d_in[10];
  const float* bo          = (const float*)d_in[11];
  const float* W_rad0      = (const float*)d_in[12];
  const float* b_rad0      = (const float*)d_in[13];
  const float* W_rad1      = (const float*)d_in[14];
  const float* b_rad1      = (const float*)d_in[15];
  const float* W_msg       = (const float*)d_in[16];
  const float* W_sh        = (const float*)d_in[17];
  const float* W_alpha     = (const float*)d_in[18];

  char* ws = (char*)d_ws;
  float* w_wkq  = (float*)(ws + 0);            // 128*16 f32        (8 KB)
  float* w_vf   = (float*)(ws + 8192);         // N*128 f32         (20 MB)
  float* w_nlog = (float*)(ws + 20979712);     // N*4 f32
  float* w_log  = (float*)(ws + 21635072);     // E*4 f32 (logits -> p)
  float* w_m    = (float*)(ws + 25829376);     // NQ*4 f32
  float* w_den  = (float*)(ws + 25894912);     // NQ*4 f32
  float* w_acc  = (float*)(ws + 25960448);     // NQ*128 f32

  k_prep<<<1, 128, 0, stream>>>(b_dst, Wq, Wk, w_wkq);
  k_init<<<(NQd * FDIM) / 256, 256, 0, stream>>>(w_m, w_den, w_acc);
  k_project<<<NNODE / 16, 256, 0, stream>>>(node_feature, Wv, w_wkq, w_vf, w_nlog);
  k_edges<<<EEDGE / 128, 256, 0, stream>>>(query_coord, node_coord, edge_src, edge_dst,
                                           scale_id, W_rad0, b_rad0, W_rad1, b_rad1,
                                           W_alpha, w_nlog, w_log, w_m);
  k_soft<<<(EEDGE * HH) / 256, 256, 0, stream>>>(edge_dst, w_m, w_log, w_den);
  k_msg<<<EEDGE / 128, 256, 0, stream>>>(query_coord, node_coord, edge_src, edge_dst,
                                         scale_id, W_rad0, b_rad0, W_rad1, b_rad1,
                                         W_msg, W_sh, w_vf, w_log, w_den, w_acc);
  k_out<<<NQd / 16, 256, 0, stream>>>(w_acc, Wo, bo, (float*)d_out);
}
